// TMP_75076028334403
// MI455X (gfx1250) — matrix operand order repaired and checked by reading, not compiled, not run on hardware
//
#include <hip/hip_runtime.h>

// ---------- CDNA5 WMMA types ----------
typedef __attribute__((ext_vector_type(16))) __bf16 v16bf;
typedef __attribute__((ext_vector_type(8)))  float  v8f;

union FragU { uint4 q[2]; v16bf v; };

__device__ __forceinline__ unsigned short f2bf(float f) {
  unsigned int u = __float_as_uint(f);
  unsigned int r = u + 0x7FFFu + ((u >> 16) & 1u);   // round-to-nearest-even
  return (unsigned short)(r >> 16);
}

// ---------- problem constants ----------
#define NN   30000
#define NE   300000
#define NG   512
#define HID  600
#define INF  9
#define NT   128
#define KP   608      // 600 padded to multiple of 32
#define LDW  640      // weight N-stride (padded)
#define MP   30208    // 30000 padded to multiple of 256 (row over-read safety)
#define LDSB 72       // LDS row stride in elements (64 + 8 pad vs bank conflicts)

// CDNA5 async global->LDS copy (ASYNCcnt) via inline asm; the clang builtin's
// pointer params live in the cuda_device/cuda_shared language address spaces,
// which are not spellable from HIP source, so we bypass it (per CDNA5_HIP.md).
#define USE_ASYNC_ASM 1

__device__ __forceinline__ void stage_b128(const unsigned short* g, unsigned short* l) {
#if USE_ASYNC_ASM
  unsigned lds_off = (unsigned)(unsigned long long)l;      // addr[31:0] == LDS offset
  unsigned long long ga = (unsigned long long)g;
  asm volatile("global_load_async_to_lds_b128 %0, %1, off"
               :
               : "v"(lds_off), "v"(ga)
               : "memory");
#else
  *(uint4*)l = *(const uint4*)g;
#endif
}

__device__ __forceinline__ void async_wait_le1() {
#if USE_ASYNC_ASM
  asm volatile("s_wait_asynccnt 0x1" ::: "memory");
#endif
}
__device__ __forceinline__ void async_wait_le0() {
#if USE_ASYNC_ASM
  asm volatile("s_wait_asynccnt 0x0" ::: "memory");
#endif
}

// ---------- simple utility kernels ----------
__global__ void copy_f32_k(const float* __restrict__ in, float* __restrict__ out, int n) {
  int i = blockIdx.x * blockDim.x + threadIdx.x;
  if (i < n) out[i] = in[i];
}

__global__ void zero_f32_k(float* __restrict__ p, int n) {
  int i = blockIdx.x * blockDim.x + threadIdx.x;
  if (i < n) p[i] = 0.0f;
}

// z[dst,f] += h[src,f] over all edges (z pre-initialized to h)
__global__ void edge_agg_k(const float* __restrict__ h, const int* __restrict__ ei,
                           float* __restrict__ z, int E, int F) {
  int i = blockIdx.x * blockDim.x + threadIdx.x;
  if (i >= E * F) return;
  int e = i / F;
  int f = i - e * F;
  int src = ei[e];
  int dst = ei[E + e];
  atomicAdd(&z[(size_t)dst * F + f], h[(size_t)src * F + f]);
}

// global add-pool: feats[batch[n],f] += h[n,f]
__global__ void pool_k(const float* __restrict__ h, const int* __restrict__ batch,
                       float* __restrict__ feats, int Nn, int F) {
  int i = blockIdx.x * blockDim.x + threadIdx.x;
  if (i >= Nn * F) return;
  int n = i / F;
  int f = i - n * F;
  atomicAdd(&feats[(size_t)batch[n] * F + f], h[(size_t)n * F + f]);
}

// fp32 [R x C] -> bf16 [Rp x Cp], zero-padded
__global__ void cvt_bf16_pad_k(const float* __restrict__ in, int R, int C,
                               unsigned short* __restrict__ out, int Rp, int Cp) {
  int i = blockIdx.x * blockDim.x + threadIdx.x;
  if (i >= Rp * Cp) return;
  int r = i / Cp;
  int c = i - r * Cp;
  out[i] = (r < R && c < C) ? f2bf(in[(size_t)r * C + c]) : (unsigned short)0;
}

// ---------- WMMA GEMM: C[MxN] = A[MxKp](bf16) * W[KpxN](bf16) + bias ----------
// block = 8 waves; wave tile = 32(M) x 64(N); block tile = 256 x 64.
// B tile (32K x 64N) staged once per block into double-buffered LDS via
// async global->LDS (ASYNCcnt); each wave gathers B fragments with 16-bit LDS loads.
__global__ void __launch_bounds__(256)
wmma_gemm_k(const unsigned short* __restrict__ A, int lda,
            const unsigned short* __restrict__ W, int ldw,
            const float* __restrict__ bias,
            float* __restrict__ out32, int ldo,
            unsigned short* __restrict__ obf, int ldb,
            int M, int N, int Kp, int relu)
{
  __shared__ __align__(16) unsigned short Bs[2][32 * LDSB];

  const int tid  = threadIdx.x;
  const int lane = tid & 31;
  const int wave = tid >> 5;
  const int m0 = (blockIdx.y * 8 + wave) * 32;
  const int n0 = blockIdx.x * 64;

  // B staging coords: 256 threads x 16B cover the 32x64 bf16 tile
  const int srow = tid >> 3;         // 0..31 (K row within tile)
  const int scol = (tid & 7) * 8;    // 0..56 (N col, 8 bf16 per thread)
  const unsigned short* gB = W + (size_t)srow * ldw + n0 + scol;
  unsigned short* lB0 = &Bs[0][srow * LDSB + scol];
  unsigned short* lB1 = &Bs[1][srow * LDSB + scol];

  // A fragment addressing (ISA 7.12.2, 16-bit A 16x32):
  // lanes 0-15: row=lane,    elems 0..7 -> K 0..7,  8..15 -> K 16..23
  // lanes16-31: row=lane-16, elems 0..7 -> K 8..15, 8..15 -> K 24..31
  const int mrow = m0 + (lane & 15);
  const int kofs = (lane < 16) ? 0 : 8;
  const unsigned short* Ap0 = A + (size_t)mrow * lda + kofs;
  const unsigned short* Ap1 = Ap0 + (size_t)16 * lda;

  v8f acc[8] = {};   // [mi*4 + j]: mi = M-half, j = 16-wide N frag

  const int nsteps = Kp >> 5;
  stage_b128(gB, lB0);                       // prefetch tile 0 -> buf 0

  for (int s = 0; s < nsteps; ++s) {
    const int kb = s << 5;
    if (s + 1 < nsteps) {
      // prefetch tile s+1 -> buf (s+1)&1; async loads complete in order,
      // so waiting <=1 guarantees tile s (the oldest) has landed.
      stage_b128(gB + (size_t)(kb + 32) * ldw, (s & 1) ? lB0 : lB1);
      async_wait_le1();
    } else {
      async_wait_le0();
    }
    __syncthreads();                         // tile s visible to all waves

    FragU a0, a1;
    a0.q[0] = *(const uint4*)(Ap0 + kb);
    a0.q[1] = *(const uint4*)(Ap0 + kb + 16);
    a1.q[0] = *(const uint4*)(Ap1 + kb);
    a1.q[1] = *(const uint4*)(Ap1 + kb + 16);

    // B fragments from LDS: lane l holds column N = l&15, elems 0..7 -> K 8h..8h+7, 8..15 -> K 16+8h..16+8h+7
    const unsigned short* bl = &Bs[s & 1][0] + ((lane >> 4) << 3) * LDSB + (lane & 15);
    FragU b0, b1, b2, b3;
#pragma unroll
    for (int e = 0; e < 16; ++e) {
      const unsigned short* p = bl + ((e & 7) + ((e >> 3) << 4)) * LDSB;
      ((unsigned short*)&b0)[e] = p[0];
      ((unsigned short*)&b1)[e] = p[16];
      ((unsigned short*)&b2)[e] = p[32];
      ((unsigned short*)&b3)[e] = p[48];
    }

    acc[0] = __builtin_amdgcn_wmma_f32_16x16x32_bf16(false, a0.v, false, b0.v, (short)0, acc[0], false, false);
    acc[1] = __builtin_amdgcn_wmma_f32_16x16x32_bf16(false, a0.v, false, b1.v, (short)0, acc[1], false, false);
    acc[2] = __builtin_amdgcn_wmma_f32_16x16x32_bf16(false, a0.v, false, b2.v, (short)0, acc[2], false, false);
    acc[3] = __builtin_amdgcn_wmma_f32_16x16x32_bf16(false, a0.v, false, b3.v, (short)0, acc[3], false, false);
    acc[4] = __builtin_amdgcn_wmma_f32_16x16x32_bf16(false, a1.v, false, b0.v, (short)0, acc[4], false, false);
    acc[5] = __builtin_amdgcn_wmma_f32_16x16x32_bf16(false, a1.v, false, b1.v, (short)0, acc[5], false, false);
    acc[6] = __builtin_amdgcn_wmma_f32_16x16x32_bf16(false, a1.v, false, b2.v, (short)0, acc[6], false, false);
    acc[7] = __builtin_amdgcn_wmma_f32_16x16x32_bf16(false, a1.v, false, b3.v, (short)0, acc[7], false, false);

    __syncthreads();                         // protect buf before re-stage
  }

  // D layout: lane l, elem i -> M = mtop + i, N = n0 + 16*j + (l&15)
  const int nc = lane & 15;
#pragma unroll
  for (int mi = 0; mi < 2; ++mi) {
    const int mtop = m0 + 16 * mi + ((lane >> 4) << 3);
#pragma unroll
    for (int j = 0; j < 4; ++j) {
      int nb = n0 + 16 * j + nc;
      float bv = (nb < N) ? bias[nb] : 0.0f;
      v8f a = acc[mi * 4 + j];
#pragma unroll
      for (int i = 0; i < 8; ++i) {
        int m = mtop + i;
        if (m < M) {
          float v = a[i] + bv;
          if (relu) v = fmaxf(v, 0.0f);
          if (out32 != nullptr && nb < N) out32[(size_t)m * ldo + nb] = v;
          if (obf != nullptr) {
            if (nb < N)        obf[(size_t)m * ldb + nb] = f2bf(v);
            else if (nb < ldb) obf[(size_t)m * ldb + nb] = 0;  // zero K padding
          }
        }
      }
    }
  }
}

// ---------- host orchestration ----------
extern "C" void kernel_launch(void* const* d_in, const int* in_sizes, int n_in,
                              void* d_out, int out_size, void* d_ws, size_t ws_size,
                              hipStream_t stream) {
  (void)in_sizes; (void)n_in; (void)out_size; (void)ws_size;

  const float* x     = (const float*)d_in[0];
  const int*   ei    = (const int*)  d_in[1];
  const int*   batch = (const int*)  d_in[2];
  const float* w1_0  = (const float*)d_in[3];
  const float* b1_0  = (const float*)d_in[4];
  const float* w2_0  = (const float*)d_in[5];
  const float* b2_0  = (const float*)d_in[6];
  const float* w1_r  = (const float*)d_in[7];
  const float* b1_r  = (const float*)d_in[8];
  const float* w2_r  = (const float*)d_in[9];
  const float* b2_r  = (const float*)d_in[10];
  const float* cw1   = (const float*)d_in[11];
  const float* cb1   = (const float*)d_in[12];
  const float* cw2   = (const float*)d_in[13];
  const float* cb2   = (const float*)d_in[14];
  const float* cw3   = (const float*)d_in[15];
  const float* cb3   = (const float*)d_in[16];
  float* out = (float*)d_out;

  // workspace carve-up
  char* ws = (char*)d_ws;
  size_t off = 0;
  auto bump = [&](size_t bytes) -> size_t {
    size_t o = off; off += (bytes + 255) & ~(size_t)255; return o;
  };
  float*          hbuf  = (float*)         (ws + bump((size_t)NN * HID * 4));
  float*          zbuf  = (float*)         (ws + bump((size_t)NN * HID * 4));
  unsigned short* Abf   = (unsigned short*)(ws + bump((size_t)MP * KP * 2));
  unsigned short* Tbf   = (unsigned short*)(ws + bump((size_t)MP * KP * 2));
  float*          feats = (float*)         (ws + bump((size_t)NG * HID * 4));
  unsigned short* fAbf  = (unsigned short*)(ws + bump((size_t)NG * KP * 2));
  unsigned short* g1    = (unsigned short*)(ws + bump((size_t)NG * 256 * 2));
  unsigned short* g2    = (unsigned short*)(ws + bump((size_t)NG * 256 * 2));
  unsigned short* wbase = (unsigned short*)(ws + bump((size_t)13 * KP * LDW * 2));
  auto wslot = [&](int i) { return wbase + (size_t)i * KP * LDW; };

  auto grid1 = [](long long n) { return dim3((unsigned)((n + 255) / 256)); };

  // convert+pad all weight matrices to bf16 slots [KP x LDW]
  auto cvtw = [&](const float* w, int R, int C, int slot) {
    cvt_bf16_pad_k<<<grid1((long long)KP * LDW), 256, 0, stream>>>(w, R, C, wslot(slot), KP, LDW);
  };
  cvtw(w1_0, INF, HID, 0);
  cvtw(w2_0, HID, HID, 1);
  for (int r = 0; r < 4; ++r) {
    cvtw(w1_r + (size_t)r * HID * HID, HID, HID, 2 + r);
    cvtw(w2_r + (size_t)r * HID * HID, HID, HID, 6 + r);
  }
  cvtw(cw1, HID, 256, 10);
  cvtw(cw2, 256, 256, 11);
  cvtw(cw3, 256, NT,  12);

  auto gemm = [&](const unsigned short* A, int lda, int slot, const float* bias,
                  float* o32, int ldo, unsigned short* obf, int ldb,
                  int M, int N, int Kp, int relu) {
    dim3 g((N + 63) / 64, (M + 255) / 256);
    wmma_gemm_k<<<g, 256, 0, stream>>>(A, lda, wslot(slot), LDW, bias,
                                       o32, ldo, obf, ldb, M, N, Kp, relu);
  };

  // ---- GIN layer 0 (in feats = 9) ----
  copy_f32_k<<<grid1((long long)NN * INF), 256, 0, stream>>>(x, zbuf, NN * INF);
  edge_agg_k<<<grid1((long long)NE * INF), 256, 0, stream>>>(x, ei, zbuf, NE, INF);
  cvt_bf16_pad_k<<<grid1((long long)NN * 32), 256, 0, stream>>>(zbuf, NN, INF, Abf, NN, 32);
  gemm(Abf, 32, 0, b1_0, nullptr, 0, Tbf, KP, NN, HID, 32, 1);          // relu(z@w1+b1) -> bf16
  gemm(Tbf, KP, 1, b2_0, hbuf, HID, nullptr, 0, NN, HID, KP, 1);        // @w2+b2, inter-layer relu

  // ---- GIN layers 1..4 ----
  for (int r = 0; r < 4; ++r) {
    copy_f32_k<<<grid1((long long)NN * HID), 256, 0, stream>>>(hbuf, zbuf, NN * HID);
    edge_agg_k<<<grid1((long long)NE * HID), 256, 0, stream>>>(hbuf, ei, zbuf, NE, HID);
    cvt_bf16_pad_k<<<grid1((long long)NN * KP), 256, 0, stream>>>(zbuf, NN, HID, Abf, NN, KP);
    gemm(Abf, KP, 2 + r, b1_r + (size_t)r * HID, nullptr, 0, Tbf, KP, NN, HID, KP, 1);
    gemm(Tbf, KP, 6 + r, b2_r + (size_t)r * HID, hbuf, HID, nullptr, 0, NN, HID, KP,
         (r < 3) ? 1 : 0);                                              // no relu after last layer
  }

  // ---- global add-pool ----
  zero_f32_k<<<grid1((long long)NG * HID), 256, 0, stream>>>(feats, NG * HID);
  pool_k<<<grid1((long long)NN * HID), 256, 0, stream>>>(hbuf, batch, feats, NN, HID);

  // ---- classifier head ----
  cvt_bf16_pad_k<<<grid1((long long)NG * KP), 256, 0, stream>>>(feats, NG, HID, fAbf, NG, KP);
  gemm(fAbf, KP, 10, cb1, nullptr, 0, g1, 256, NG, 256, KP,  1);
  gemm(g1,  256, 11, cb2, nullptr, 0, g2, 256, NG, 256, 256, 1);
  gemm(g2,  256, 12, cb3, out, NT, nullptr, 0, NG, NT,  256, 0);
}
